// EncoderAttention_18064632447447
// MI455X (gfx1250) — matrix operand order repaired and checked by reading, not compiled, not run on hardware
//
#include <hip/hip_runtime.h>
#include <hip/hip_bf16.h>

typedef __attribute__((ext_vector_type(16))) _Float16 v16h;
typedef __attribute__((ext_vector_type(8)))  float    v8f;

#define BN     4
#define CC     128
#define NHEAD  4
#define NTOK   64
#define HDIM   32
#define MLPH   512
#define HWTOT  16384
#define TOKS   (BN*HWTOT)      // 65536
#define ROWS   (2*TOKS)        // 131072 (x-branch rows then d-branch rows)
#define NWIN   1024            // windows per branch
#define QSCALE 0.17677669529663687f   // 32^-0.5

#define LDA_S  40   // A LDS row stride (halves): 80B, 16B-divisible
#define LDBT_S 40   // B LDS row stride (halves)
#define LDVT_S 72   // V LDS row stride (halves): 144B, 16B-divisible
#define LDP_S  72

// -------- CDNA5 async global->LDS (ASYNCcnt) with safe fallback ----------
#if defined(__has_builtin)
# if __has_builtin(__builtin_amdgcn_global_load_async_to_lds_b128) && \
     __has_builtin(__builtin_amdgcn_s_wait_asynccnt)
#  define HAVE_ASYNC_LDS 1
# endif
#endif
#ifndef HAVE_ASYNC_LDS
# define HAVE_ASYNC_LDS 0
#endif

#if HAVE_ASYNC_LDS
typedef int v4i_ __attribute__((__vector_size__(4 * sizeof(int))));
typedef __attribute__((address_space(1))) v4i_ gv4i;   // global int4
typedef __attribute__((address_space(3))) v4i_ lv4i;   // LDS int4
#endif

// copy 32 bytes (2 x b128) global -> LDS at per-thread addresses
__device__ __forceinline__ void cp32_g2l(const _Float16* g, _Float16* l) {
#if HAVE_ASYNC_LDS
  gv4i* gp = (gv4i*)(void*)g;
  lv4i* lp = (lv4i*)(void*)l;
  __builtin_amdgcn_global_load_async_to_lds_b128(gp, lp, 0, 0);
  __builtin_amdgcn_global_load_async_to_lds_b128(gp, lp, 16, 0);
#else
  const uint4* s = (const uint4*)g;
  uint4* d = (uint4*)l;
  d[0] = s[0];
  d[1] = s[1];
#endif
}
__device__ __forceinline__ void async_drain() {
#if HAVE_ASYNC_LDS
  __builtin_amdgcn_s_wait_asynccnt(0);
#endif
}

// ---------------------------------------------------------------- utilities

// Emit weight in k-tiled, fragment-ready layout:
//   Wt[((k/32) * Nn + n) * 32 + (k%32)] = W[n][k]
// so a 64x32 GEMM B-tile is one contiguous 4KB block, n-major, 32 halves/n.
__global__ void k_cvt_w(const float* __restrict__ W, _Float16* __restrict__ Wt,
                        int Nn, int Kk) {
  int i = blockIdx.x * 256 + threadIdx.x;
  if (i >= Nn * Kk) return;
  int n = i / Kk, k = i % Kk;
  Wt[((size_t)(k >> 5) * Nn + n) * 32 + (k & 31)] = (_Float16)W[(size_t)n * Kk + k];
}

// One wave (32 lanes) per 128-wide row; 8 rows per 256-thread block.
__global__ __launch_bounds__(256) void k_layernorm(
    const float* __restrict__ x, const float* __restrict__ g,
    const float* __restrict__ b, _Float16* __restrict__ out, int rows) {
  int row  = blockIdx.x * 8 + (threadIdx.x >> 5);
  int lane = threadIdx.x & 31;
  if (row >= rows) return;
  float4 v4 = ((const float4*)(x + (size_t)row * CC))[lane];
  float v[4] = {v4.x, v4.y, v4.z, v4.w};
  float s = v[0] + v[1] + v[2] + v[3];
  #pragma unroll
  for (int m = 16; m >= 1; m >>= 1) s += __shfl_xor(s, m, 32);
  float mean = s * (1.0f / CC);
  float q = 0.f;
  #pragma unroll
  for (int i = 0; i < 4; i++) { float d = v[i] - mean; q += d * d; }
  #pragma unroll
  for (int m = 16; m >= 1; m >>= 1) q += __shfl_xor(q, m, 32);
  float rstd = rsqrtf(q * (1.0f / CC) + 1e-5f);
  _Float16* o = out + (size_t)row * CC + lane * 4;
  #pragma unroll
  for (int i = 0; i < 4; i++) {
    int c = lane * 4 + i;
    o[i] = (_Float16)((v[i] - mean) * rstd * g[c] + b[c]);
  }
}

// ------------------------------------------------- 64x64 WMMA GEMM core
// A: (M,K) f16 row-major. Bt: weight in k-tiled layout (see k_cvt_w), total
// width Nn. 128 threads = 4 waves; wave w computes rows [64*bx+16w,+16) x 64.
// Double-buffered LDS; BOTH tiles staged with async global->LDS b128 copies
// (no VGPR bounce, no LDS scatter); one barrier per k-step. All WMMA
// fragments are contiguous 16-half runs -> paired ds_load_b128.
template <int K>
__device__ __forceinline__ void gemm_core(
    const _Float16* __restrict__ A, int lda, int m0g,
    const _Float16* __restrict__ Bt, int Nn, int n0g,
    _Float16* sA, _Float16* sBt, v8f acc[4]) {
  int tid  = threadIdx.x;
  int lane = tid & 31;
  int wave = tid >> 5;
  #pragma unroll
  for (int t = 0; t < 4; t++) acc[t] = (v8f){};
  int kh = (lane >> 4) << 4;          // A K-half select (0 or 16)
  int kb = (lane >> 4) << 4;          // B K-half select (0 or 16)
  int ar_ = tid >> 1, ac_ = (tid & 1) * 16;    // A staging coords
  int bn_ = tid >> 1, bh_ = (tid & 1) * 16;    // B staging coords
  const int ASZ = 64 * LDA_S, BSZ = 64 * LDBT_S;

  auto stage = [&](int buf, int k0) {
    cp32_g2l(A + (size_t)(m0g + ar_) * lda + k0 + ac_,
             sA + buf * ASZ + ar_ * LDA_S + ac_);
    cp32_g2l(Bt + ((size_t)(k0 >> 5) * Nn + n0g + bn_) * 32 + bh_,
             sBt + buf * BSZ + bn_ * LDBT_S + bh_);
  };

  stage(0, 0);
  #pragma unroll
  for (int k0 = 0; k0 < K; k0 += 32) {
    int buf = (k0 >> 5) & 1;
    async_drain();       // this wave's async LDS writes for `buf` are done
    __syncthreads();     // everyone's tile `buf` visible; prior reads retired
    if (k0 + 32 < K) stage(buf ^ 1, k0 + 32);   // overlap with compute
    if (k0 + 64 < K)
      __builtin_prefetch(Bt + ((size_t)((k0 >> 5) + 2) * Nn + n0g + bn_) * 32 + bh_, 0, 1);
    const _Float16* cA = sA + buf * ASZ;
    const _Float16* cB = sBt + buf * BSZ;
    v16h a;
    { // A fragment: one contiguous 16-half run (k = 16*half + e)
      int m = (wave << 4) + (lane & 15);
      const _Float16* ar = cA + m * LDA_S + kh;
      #pragma unroll
      for (int e = 0; e < 16; e++) a[e] = ar[e];
    }
    #pragma unroll
    for (int t = 0; t < 4; t++) {
      v16h bfrag;   // 16 contiguous halves from B row n
      const _Float16* br = cB + ((t << 4) + (lane & 15)) * LDBT_S + kb;
      #pragma unroll
      for (int e = 0; e < 16; e++) bfrag[e] = br[e];
      acc[t] = __builtin_amdgcn_wmma_f32_16x16x32_f16(
          false, a, false, bfrag, (short)0, acc[t], false, false);
    }
  }
}

// ------------------------------------------------------------- QKV GEMM
// V chunk is written PRE-TRANSPOSED ([hd][token] per window-head) so the
// attention kernel can stage it with pure async-to-LDS copies.
__global__ __launch_bounds__(128) void k_gemm_qkv(
    const _Float16* __restrict__ A, const _Float16* __restrict__ Bt,
    const float* __restrict__ bias,
    _Float16* __restrict__ Qb, _Float16* __restrict__ Kb,
    _Float16* __restrict__ Vb, _Float16* __restrict__ Q2b) {
  __shared__ __align__(16) _Float16 sA[2 * 64 * LDA_S];
  __shared__ __align__(16) _Float16 sBt[2 * 64 * LDBT_S];
  v8f acc[4];
  gemm_core<CC>(A, CC, blockIdx.x * 64, Bt, 4 * CC, blockIdx.y * 64, sA, sBt, acc);
  int lane = threadIdx.x & 31, wave = threadIdx.x >> 5;
  int chunk = blockIdx.y >> 1;                   // uniform per block
  float sc = (chunk == 0 || chunk == 3) ? QSCALE : 1.0f;
  bool isV = (chunk == 2);
  _Float16* dst = (chunk == 0) ? Qb : (chunk == 1) ? Kb : (chunk == 2) ? Vb : Q2b;
  #pragma unroll
  for (int t = 0; t < 4; t++) {
    #pragma unroll
    for (int r = 0; r < 8; r++) {
      int gm = blockIdx.x * 64 + (wave << 4) + ((lane >> 4) << 3) + r;
      int gn = blockIdx.y * 64 + (t << 4) + (lane & 15);
      float v = (acc[t][r] + bias[gn]) * sc;
      int branch = gm >> 16, tok = gm & 65535;
      int b = tok >> 14, hw = tok & 16383;
      int i = hw >> 7, j = hw & 127;
      int win = branch * NWIN + b * 256 + ((i >> 3) << 4) + (j >> 3);
      int pos = ((i & 7) << 3) + (j & 7);
      int h = (gn >> 5) & 3, hd = gn & 31;
      size_t whbase = (size_t)(win * NHEAD + h) * (NTOK * HDIM);
      size_t idx = whbase + (isV ? (size_t)(hd * NTOK + pos)
                                 : (size_t)(pos * HDIM + hd));
      dst[idx] = (_Float16)v;
    }
  }
}

// ---------------------------------------------------------- attention
// One block per (branch, window, head). S1=Q*K^T (+bias on branch 0),
// S2=Q2*K^T with Q2 from the opposite branch; O=(softmax(S1)+softmax(S2))*V.
__global__ __launch_bounds__(128) void k_attn(
    const _Float16* __restrict__ Qb, const _Float16* __restrict__ Kb,
    const _Float16* __restrict__ Vb, const _Float16* __restrict__ Q2b,
    const float* __restrict__ rpb, _Float16* __restrict__ Ob) {
  __shared__ __align__(16) _Float16 sVt[HDIM * LDVT_S];     // V: [hd][token]
  __shared__ __align__(16) _Float16 sP[4 * 16 * LDP_S];
  int id = blockIdx.x;
  int branch = id >> 12;
  int win = (id >> 2) & 1023;
  int head = id & 3;
  int lane = threadIdx.x & 31;
  int wave = threadIdx.x >> 5;
  size_t base  = ((size_t)((branch * NWIN + win) * NHEAD + head)) * (NTOK * HDIM);
  size_t base2 = ((size_t)(((1 - branch) * NWIN + win) * NHEAD + head)) * (NTOK * HDIM);
  const _Float16* Qg  = Qb  + base;
  const _Float16* Kg  = Kb  + base;
  const _Float16* Vg  = Vb  + base;   // already [hd][token]
  const _Float16* Q2g = Q2b + base2;
  { // stage V with async global->LDS: 32B per thread, already transposed
    int hd_ = threadIdx.x >> 2, q_ = (threadIdx.x & 3) * 16;
    cp32_g2l(Vg + hd_ * NTOK + q_, sVt + hd_ * LDVT_S + q_);
  }
  int m0 = wave << 4;
  int kh = (lane >> 4) << 4;
  v16h aq, aq2;
  { // A fragments: this wave's 16 query rows, K=0..31 (=HDIM)
    int m = m0 + (lane & 15);
    const _Float16* q1 = Qg  + m * HDIM + kh;
    const _Float16* q2 = Q2g + m * HDIM + kh;
    #pragma unroll
    for (int e = 0; e < 16; e++) {
      aq[e] = q1[e];
      aq2[e] = q2[e];
    }
  }
  v8f s1[4], s2[4];
  #pragma unroll
  for (int t = 0; t < 4; t++) {
    v16h bk;   // B = K^T: element e -> Kmat[n][(lanehalf*16)+e], contiguous row read
    int n = (t << 4) + (lane & 15);
    const _Float16* kr = Kg + n * HDIM + ((lane >> 4) << 4);
    #pragma unroll
    for (int e = 0; e < 16; e++) bk[e] = kr[e];
    v8f z = (v8f){};
    s1[t] = __builtin_amdgcn_wmma_f32_16x16x32_f16(false, aq,  false, bk, (short)0, z, false, false);
    s2[t] = __builtin_amdgcn_wmma_f32_16x16x32_f16(false, aq2, false, bk, (short)0, z, false, false);
  }
  if (branch == 0) { // relative position bias only on the r-branch primary attn
    #pragma unroll
    for (int t = 0; t < 4; t++) {
      #pragma unroll
      for (int r = 0; r < 8; r++) {
        int qpos = m0 + ((lane >> 4) << 3) + r;
        int kpos = (t << 4) + (lane & 15);
        int idx = ((qpos >> 3) - (kpos >> 3) + 7) * 15 + ((qpos & 7) - (kpos & 7) + 7);
        s1[t][r] += rpb[idx * NHEAD + head];
      }
    }
  }
  // dual row softmax over 64 keys (row lives in one 16-lane half x 4 tiles),
  // write P1+P2 as f16 A-operand staging.
  _Float16* sPw = sP + wave * (16 * LDP_S);
  #pragma unroll
  for (int r = 0; r < 8; r++) {
    float mx1 = -3.4e38f, mx2 = -3.4e38f;
    #pragma unroll
    for (int t = 0; t < 4; t++) { mx1 = fmaxf(mx1, s1[t][r]); mx2 = fmaxf(mx2, s2[t][r]); }
    #pragma unroll
    for (int m = 8; m >= 1; m >>= 1) {
      mx1 = fmaxf(mx1, __shfl_xor(mx1, m, 32));
      mx2 = fmaxf(mx2, __shfl_xor(mx2, m, 32));
    }
    float e1[4], e2[4], sm1 = 0.f, sm2 = 0.f;
    #pragma unroll
    for (int t = 0; t < 4; t++) {
      e1[t] = __expf(s1[t][r] - mx1); sm1 += e1[t];
      e2[t] = __expf(s2[t][r] - mx2); sm2 += e2[t];
    }
    #pragma unroll
    for (int m = 8; m >= 1; m >>= 1) {
      sm1 += __shfl_xor(sm1, m, 32);
      sm2 += __shfl_xor(sm2, m, 32);
    }
    float i1 = 1.0f / sm1, i2 = 1.0f / sm2;
    int row = ((lane >> 4) << 3) + r;
    #pragma unroll
    for (int t = 0; t < 4; t++)
      sPw[row * LDP_S + (t << 4) + (lane & 15)] = (_Float16)(e1[t] * i1 + e2[t] * i2);
  }
  async_drain();     // V async-to-LDS writes complete before the barrier
  __syncthreads();
  // O = P * V : 16x32 per wave, K=64 in two steps; V fragments contiguous in sVt
  v8f o[2] = {(v8f){}, (v8f){}};
  #pragma unroll
  for (int ks = 0; ks < 2; ks++) {
    v16h ap;
    {
      int m = lane & 15;
      const _Float16* pr = sPw + m * LDP_S + ks * 32 + kh;
      #pragma unroll
      for (int e = 0; e < 16; e++) ap[e] = pr[e];
    }
    #pragma unroll
    for (int t = 0; t < 2; t++) {
      v16h bv;
      int n = (t << 4) + (lane & 15);
      const _Float16* vr = sVt + n * LDVT_S + ks * 32 + ((lane >> 4) << 4);
      #pragma unroll
      for (int e = 0; e < 16; e++) bv[e] = vr[e];
      o[t] = __builtin_amdgcn_wmma_f32_16x16x32_f16(false, ap, false, bv, (short)0, o[t], false, false);
    }
  }
  size_t orow = (size_t)(branch * NWIN + win) * NTOK;
  #pragma unroll
  for (int t = 0; t < 2; t++) {
    #pragma unroll
    for (int r = 0; r < 8; r++) {
      int qpos = m0 + ((lane >> 4) << 3) + r;
      int hd = (t << 4) + (lane & 15);
      Ob[(orow + qpos) * CC + head * HDIM + hd] = (_Float16)o[t][r];
    }
  }
}

// ------------------------------------------------- proj GEMM + residual
__global__ __launch_bounds__(128) void k_gemm_proj(
    const _Float16* __restrict__ A, const _Float16* __restrict__ Bt,
    const float* __restrict__ bias, const float* __restrict__ xin,
    const float* __restrict__ din, float* __restrict__ xr) {
  __shared__ __align__(16) _Float16 sA[2 * 64 * LDA_S];
  __shared__ __align__(16) _Float16 sBt[2 * 64 * LDBT_S];
  v8f acc[4];
  gemm_core<CC>(A, CC, blockIdx.x * 64, Bt, CC, blockIdx.y * 64, sA, sBt, acc);
  int lane = threadIdx.x & 31, wave = threadIdx.x >> 5;
  #pragma unroll
  for (int t = 0; t < 4; t++) {
    #pragma unroll
    for (int r = 0; r < 8; r++) {
      int gm = blockIdx.x * 64 + (wave << 4) + ((lane >> 4) << 3) + r;
      int gn = blockIdx.y * 64 + (t << 4) + (lane & 15);
      int branch = gm >> 16, rem = gm & 65535;
      int win = rem >> 6, pos = rem & 63;
      int b = win >> 8, wl = win & 255;
      int i = ((wl >> 4) << 3) + (pos >> 3);
      int j = ((wl & 15) << 3) + (pos & 7);
      size_t hw = (size_t)i * 128 + j;
      const float* res = (branch == 0) ? xin : din;
      float v = acc[t][r] + bias[gn] + res[((size_t)b * HWTOT + hw) * CC + gn];
      xr[((size_t)branch * TOKS + (size_t)b * HWTOT + hw) * CC + gn] = v;
    }
  }
}

// ---------------------------------------------------- fc1 GEMM + GELU
__global__ __launch_bounds__(128) void k_gemm_fc1(
    const _Float16* __restrict__ A, const _Float16* __restrict__ Bt,
    const float* __restrict__ bias, _Float16* __restrict__ hid) {
  __shared__ __align__(16) _Float16 sA[2 * 64 * LDA_S];
  __shared__ __align__(16) _Float16 sBt[2 * 64 * LDBT_S];
  v8f acc[4];
  gemm_core<CC>(A, CC, blockIdx.x * 64, Bt, MLPH, blockIdx.y * 64, sA, sBt, acc);
  int lane = threadIdx.x & 31, wave = threadIdx.x >> 5;
  #pragma unroll
  for (int t = 0; t < 4; t++) {
    #pragma unroll
    for (int r = 0; r < 8; r++) {
      int gm = blockIdx.x * 64 + (wave << 4) + ((lane >> 4) << 3) + r;
      int gn = blockIdx.y * 64 + (t << 4) + (lane & 15);
      float v = acc[t][r] + bias[gn];
      float g = 0.5f * v * (1.0f + erff(v * 0.70710678118654752f));  // exact GELU
      hid[(size_t)gm * MLPH + gn] = (_Float16)g;
    }
  }
}

// ------------------------------------------- fc2 GEMM + residual -> out
__global__ __launch_bounds__(128) void k_gemm_fc2(
    const _Float16* __restrict__ A, const _Float16* __restrict__ Bt,
    const float* __restrict__ bias, const float* __restrict__ xr,
    float* __restrict__ out) {
  __shared__ __align__(16) _Float16 sA[2 * 64 * LDA_S];
  __shared__ __align__(16) _Float16 sBt[2 * 64 * LDBT_S];
  v8f acc[4];
  gemm_core<MLPH>(A, MLPH, blockIdx.x * 64, Bt, CC, blockIdx.y * 64, sA, sBt, acc);
  int lane = threadIdx.x & 31, wave = threadIdx.x >> 5;
  #pragma unroll
  for (int t = 0; t < 4; t++) {
    #pragma unroll
    for (int r = 0; r < 8; r++) {
      int gm = blockIdx.x * 64 + (wave << 4) + ((lane >> 4) << 3) + r;
      int gn = blockIdx.y * 64 + (t << 4) + (lane & 15);
      size_t o = (size_t)gm * CC + gn;
      out[o] = xr[o] + acc[t][r] + bias[gn];
    }
  }
}

// ------------------------------------------------------------------ host
extern "C" void kernel_launch(void* const* d_in, const int* in_sizes, int n_in,
                              void* d_out, int out_size, void* d_ws, size_t ws_size,
                              hipStream_t stream) {
  (void)in_sizes; (void)n_in; (void)out_size; (void)ws_size;
  const float* x     = (const float*)d_in[0];
  const float* d     = (const float*)d_in[1];
  const float* rpb   = (const float*)d_in[2];
  const float* n1w   = (const float*)d_in[3];
  const float* n1b   = (const float*)d_in[4];
  const float* n2w   = (const float*)d_in[5];
  const float* n2b   = (const float*)d_in[6];
  const float* qkvw  = (const float*)d_in[7];
  const float* qkvb  = (const float*)d_in[8];
  const float* projw = (const float*)d_in[9];
  const float* projb = (const float*)d_in[10];
  const float* fc1w  = (const float*)d_in[11];
  const float* fc1b  = (const float*)d_in[12];
  const float* fc2w  = (const float*)d_in[13];
  const float* fc2b  = (const float*)d_in[14];
  float* out = (float*)d_out;

  char* ws = (char*)d_ws;
  size_t off = 0;
  auto alloc = [&](size_t bytes) {
    void* p = ws + off;
    off += (bytes + 255) & ~(size_t)255;
    return p;
  };
  _Float16* ln1    = (_Float16*)alloc((size_t)ROWS * CC * 2);
  _Float16* qkvWt  = (_Float16*)alloc((size_t)CC * 4 * CC * 2);
  _Float16* projWt = (_Float16*)alloc((size_t)CC * CC * 2);
  _Float16* fc1Wt  = (_Float16*)alloc((size_t)CC * MLPH * 2);
  _Float16* fc2Wt  = (_Float16*)alloc((size_t)MLPH * CC * 2);
  size_t qsz = (size_t)2 * NWIN * NHEAD * NTOK * HDIM * 2;  // 32 MiB, 256B-aligned
  _Float16* Qb  = (_Float16*)alloc(qsz);
  _Float16* Kb  = (_Float16*)alloc(qsz);
  _Float16* Vb  = (_Float16*)alloc(qsz);
  _Float16* Q2b = (_Float16*)alloc(qsz);
  _Float16* Ob  = (_Float16*)alloc((size_t)ROWS * CC * 2);
  float*    xr  = (float*)alloc((size_t)ROWS * CC * 4);
  _Float16* ln2    = ln1;  // LN1 buffer dead after QKV
  _Float16* hidden = Qb;   // Q/K/V/Q2 contiguous 4*qsz == ROWS*MLPH*2, dead after attn

  // weights -> f16, k-tiled fragment-ready layout
  int nw = 4 * CC * CC;
  k_cvt_w<<<(nw + 255) / 256, 256, 0, stream>>>(qkvw, qkvWt, 4 * CC, CC);
  k_cvt_w<<<(CC * CC + 255) / 256, 256, 0, stream>>>(projw, projWt, CC, CC);
  k_cvt_w<<<(nw + 255) / 256, 256, 0, stream>>>(fc1w, fc1Wt, MLPH, CC);
  k_cvt_w<<<(nw + 255) / 256, 256, 0, stream>>>(fc2w, fc2Wt, CC, MLPH);
  // LN1 for both branches
  k_layernorm<<<TOKS / 8, 256, 0, stream>>>(x, n1w, n1b, ln1, TOKS);
  k_layernorm<<<TOKS / 8, 256, 0, stream>>>(d, n1w, n1b, ln1 + (size_t)TOKS * CC, TOKS);
  // QKV
  k_gemm_qkv<<<dim3(ROWS / 64, 8), 128, 0, stream>>>(ln1, qkvWt, qkvb, Qb, Kb, Vb, Q2b);
  // windowed dual attention
  k_attn<<<2 * NWIN * NHEAD, 128, 0, stream>>>(Qb, Kb, Vb, Q2b, rpb, Ob);
  // proj + residual (window reverse folded into epilogue)
  k_gemm_proj<<<dim3(ROWS / 64, 2), 128, 0, stream>>>(Ob, projWt, projb, x, d, xr);
  // LN2
  k_layernorm<<<ROWS / 8, 256, 0, stream>>>(xr, n2w, n2b, ln2, ROWS);
  // MLP
  k_gemm_fc1<<<dim3(ROWS / 64, 8), 128, 0, stream>>>(ln2, fc1Wt, fc1b, hidden);
  k_gemm_fc2<<<dim3(ROWS / 64, 2), 128, 0, stream>>>(hidden, fc2Wt, fc2b, xr, out);
}
